// AdaptiveMonolithGNN_55241869361653
// MI455X (gfx1250) — compile-verified, matrix operand order checked by reading, not run on hardware
//
#include <hip/hip_runtime.h>
#include <stdint.h>

#define HD __device__ __forceinline__

typedef __attribute__((ext_vector_type(16))) __bf16 v16bf;
typedef __attribute__((ext_vector_type(8)))  float  v8f;

static const int H_NHEADS[5] = {8, 7, 5, 3, 1};
static const int H_DIN[5]    = {128, 256, 224, 160, 96};
static const int H_DOUT[5]   = {256, 224, 160, 96, 32};
#define BN_EPS 1e-5f
#define NEG_SLOPE 0.2f

// ---------- small device helpers ----------
HD unsigned int f2bf(float f) {             // f32 -> bf16 RNE (returned in low 16)
  unsigned int u = __float_as_uint(f);
  u += 0x7FFFu + ((u >> 16) & 1u);
  return u >> 16;
}
HD unsigned int f2ord(float f) {            // order-preserving float->uint
  unsigned int u = __float_as_uint(f);
  return (u & 0x80000000u) ? ~u : (u | 0x80000000u);
}
HD float ord2f(unsigned int u) {
  return __uint_as_float((u & 0x80000000u) ? (u & 0x7FFFFFFFu) : ~u);
}
HD float leaky(float v) { return v > 0.f ? v : NEG_SLOPE * v; }

// ---------- utility ----------
__global__ void zero_kernel(float* __restrict__ p, int n) {
  int t = blockIdx.x * blockDim.x + threadIdx.x;
  if (t < n) p[t] = 0.f;
}

// ---------- fragment packing ----------
// Packs a row-major f32 matrix into WMMA-fragment-major bf16 records:
// record index = ((tile * ksteps + ks) * 32 + lane), 8 dwords per record.
// 16-bit A 16x32 layout: K = ks*32 + (j>>2)*16 + (lane>>4)*8 + (j&3)*2.
__global__ void pack_a_kernel(const float* __restrict__ X, unsigned int* __restrict__ Ap,
                              int M, int K) {
  int t = blockIdx.x * blockDim.x + threadIdx.x;
  int ksteps = K >> 5;
  int total = (M >> 4) * ksteps * 256;
  if (t >= total) return;
  int j = t & 7, lane = (t >> 3) & 31;
  int ks = (t >> 8) % ksteps, rt = (t >> 8) / ksteps;
  int half = lane >> 4, m16 = lane & 15;
  int kk = (ks << 5) + ((j >> 2) << 4) + (half << 3) + ((j & 3) << 1);
  size_t base = (size_t)(rt * 16 + m16) * K + kk;
  Ap[t] = f2bf(X[base]) | (f2bf(X[base + 1]) << 16);
}

// B (K x Nd): tile over Nd (col = tile*16 + lane&15), same per-lane K schedule.
__global__ void pack_b_kernel(const float* __restrict__ W, unsigned int* __restrict__ Bp,
                              int K, int Nd) {
  int t = blockIdx.x * blockDim.x + threadIdx.x;
  int ksteps = K >> 5;
  int total = (Nd >> 4) * ksteps * 256;
  if (t >= total) return;
  int j = t & 7, lane = (t >> 3) & 31;
  int ks = (t >> 8) % ksteps, ct = (t >> 8) / ksteps;
  int half = lane >> 4, m16 = lane & 15;
  int kk = (ks << 5) + ((j >> 2) << 4) + (half << 3) + ((j & 3) << 1);
  int col = (ct << 4) + m16;
  Bp[t] = f2bf(W[(size_t)kk * Nd + col]) | (f2bf(W[(size_t)(kk + 1) * Nd + col]) << 16);
}

// ---------- WMMA GEMM: C[M,Nd] = A x B, bf16 in / f32 acc ----------
// One wave32 computes a 64x16 block (4 row-tiles x 1 col-tile), reusing the
// B fragment across 4 v_wmma per k-step. Branch-free k-loop: remainder block
// clamps duplicate row-tiles for loads; only stores are guarded (uniform).
union Frag { v16bf v; uint4 q[2]; };

HD void store_tile(float* __restrict__ C, const v8f& a, int rowBase, int col, int Nd) {
  size_t cb = (size_t)rowBase * Nd + col;
#pragma unroll
  for (int r = 0; r < 8; ++r) C[cb + (size_t)r * Nd] = a[r];
}

__global__ void __launch_bounds__(256)
gemm_wmma4(const unsigned int* __restrict__ Ap, const unsigned int* __restrict__ Bp,
           float* __restrict__ C, int M, int K, int Nd) {
  int nColTiles = Nd >> 4;
  int nRowTiles = M >> 4;
  int ksteps = K >> 5;
  int nRowBlk = (nRowTiles + 3) >> 2;
  int wave = blockIdx.x * (blockDim.x >> 5) + (threadIdx.x >> 5);
  if (wave >= nRowBlk * nColTiles) return;   // wave-uniform exit
  int colTile = wave % nColTiles;
  int rt0 = (wave / nColTiles) << 2;
  int lane = threadIdx.x & 31;
  int half = lane >> 4, m16 = lane & 15;

  int r1 = rt0 + 1 < nRowTiles ? rt0 + 1 : nRowTiles - 1;
  int r2 = rt0 + 2 < nRowTiles ? rt0 + 2 : nRowTiles - 1;
  int r3 = rt0 + 3 < nRowTiles ? rt0 + 3 : nRowTiles - 1;

  // per-lane fragment pointers; +64 uint4 (1KB) per k-step
  const uint4* bp  = (const uint4*)Bp + (size_t)colTile * ksteps * 64 + lane * 2;
  const uint4* ap0 = (const uint4*)Ap + (size_t)rt0 * ksteps * 64 + lane * 2;
  const uint4* ap1 = (const uint4*)Ap + (size_t)r1  * ksteps * 64 + lane * 2;
  const uint4* ap2 = (const uint4*)Ap + (size_t)r2  * ksteps * 64 + lane * 2;
  const uint4* ap3 = (const uint4*)Ap + (size_t)r3  * ksteps * 64 + lane * 2;

  v8f acc0 = (v8f){0.f,0.f,0.f,0.f,0.f,0.f,0.f,0.f};
  v8f acc1 = acc0, acc2 = acc0, acc3 = acc0;

  for (int ks = 0; ks < ksteps; ++ks) {
    Frag b_, a0, a1, a2, a3;
    b_.q[0] = bp[0];  b_.q[1] = bp[1];
    a0.q[0] = ap0[0]; a0.q[1] = ap0[1];
    a1.q[0] = ap1[0]; a1.q[1] = ap1[1];
    a2.q[0] = ap2[0]; a2.q[1] = ap2[1];
    a3.q[0] = ap3[0]; a3.q[1] = ap3[1];
    acc0 = __builtin_amdgcn_wmma_f32_16x16x32_bf16(false, a0.v, false, b_.v, (short)0, acc0, false, false);
    acc1 = __builtin_amdgcn_wmma_f32_16x16x32_bf16(false, a1.v, false, b_.v, (short)0, acc1, false, false);
    acc2 = __builtin_amdgcn_wmma_f32_16x16x32_bf16(false, a2.v, false, b_.v, (short)0, acc2, false, false);
    acc3 = __builtin_amdgcn_wmma_f32_16x16x32_bf16(false, a3.v, false, b_.v, (short)0, acc3, false, false);
    bp += 64; ap0 += 64; ap1 += 64; ap2 += 64; ap3 += 64;
  }

  // C 16x16 f32 layout: element (Mrow = 8*half + r, Ncol = lane&15) in VGPR r
  int col = (colTile << 4) + m16;
  int rb = rt0 * 16 + half * 8;
  store_tile(C, acc0, rb, col, Nd);
  if (rt0 + 1 < nRowTiles) store_tile(C, acc1, rb + 16, col, Nd);
  if (rt0 + 2 < nRowTiles) store_tile(C, acc2, rb + 32, col, Nd);
  if (rt0 + 3 < nRowTiles) store_tile(C, acc3, rb + 48, col, Nd);
}

// ---------- attention coefficients: e_s[n,h], e_d[n,h] ----------
__global__ void attn_coef_kernel(const float* __restrict__ xw,
                                 const float* __restrict__ a_src,
                                 const float* __restrict__ a_dst,
                                 float* __restrict__ es, float* __restrict__ ed,
                                 int N, int H) {
  int t = blockIdx.x * blockDim.x + threadIdx.x;
  if (t >= N * H) return;
  int h = t % H;
  const float* row = xw + (size_t)t * 32;   // t == n*H + h, xw row-major [N, H*32]
  float ss = 0.f, sd = 0.f;
#pragma unroll
  for (int c = 0; c < 32; ++c) {
    float xv = row[c];
    ss = fmaf(xv, a_src[h * 32 + c], ss);
    sd = fmaf(xv, a_dst[h * 32 + c], sd);
  }
  es[t] = ss;
  ed[t] = sd;
}

// ---------- edge phase A: segment max via order-preserving uint atomicMax ----------
__global__ void edge_max_kernel(const int* __restrict__ esrc, const int* __restrict__ edst,
                                const float* __restrict__ es, const float* __restrict__ ed,
                                unsigned int* __restrict__ mord, int E, int N, int H) {
  int t = blockIdx.x * blockDim.x + threadIdx.x;
  if (t >= (E + N) * H) return;
  int h = t % H;
  int e = t / H;
  int s = (e < E) ? esrc[e] : (e - E);      // self-loops appended
  int d = (e < E) ? edst[e] : (e - E);
  float v = leaky(es[s * H + h] + ed[d * H + h]);
  atomicMax(&mord[d * H + h], f2ord(v));
}

// ---------- edge phase B: denom[dst,h] += exp(e - m[dst,h]) ----------
__global__ void edge_sum_kernel(const int* __restrict__ esrc, const int* __restrict__ edst,
                                const float* __restrict__ es, const float* __restrict__ ed,
                                const unsigned int* __restrict__ mord,
                                float* __restrict__ denom, int E, int N, int H) {
  int t = blockIdx.x * blockDim.x + threadIdx.x;
  if (t >= (E + N) * H) return;
  int h = t % H;
  int e = t / H;
  int s = (e < E) ? esrc[e] : (e - E);
  int d = (e < E) ? edst[e] : (e - E);
  int di = d * H + h;
  float v = leaky(es[s * H + h] + ed[di]);
  atomicAdd(&denom[di], __expf(v - ord2f(mord[di])));
}

// ---------- edge phase C: out[dst,h,c] += alpha * xw[src,h,c] ----------
// 8 lanes per (edge,head), float4 gather + 4 contiguous f32 atomics per lane.
__global__ void edge_scatter_kernel(const int* __restrict__ esrc, const int* __restrict__ edst,
                                    const float* __restrict__ es, const float* __restrict__ ed,
                                    const unsigned int* __restrict__ mord,
                                    const float* __restrict__ denom,
                                    const float* __restrict__ xw,
                                    float* __restrict__ out, int E, int N, int H) {
  int t = blockIdx.x * blockDim.x + threadIdx.x;
  if (t >= (E + N) * H * 8) return;
  int q = t & 7;
  int eh = t >> 3;
  int h = eh % H;
  int e = eh / H;
  int s = (e < E) ? esrc[e] : (e - E);
  int d = (e < E) ? edst[e] : (e - E);
  int di = d * H + h;
  float v = leaky(es[s * H + h] + ed[di]);
  float alpha = __expf(v - ord2f(mord[di])) / denom[di];
  float4 xv = *(const float4*)(xw + ((size_t)(s * H + h) << 5) + (q << 2));
  float* o = out + ((size_t)di << 5) + (q << 2);
  atomicAdd(o + 0, xv.x * alpha);
  atomicAdd(o + 1, xv.y * alpha);
  atomicAdd(o + 2, xv.z * alpha);
  atomicAdd(o + 3, xv.w * alpha);
}

// ---------- BN stats: thread = channel, register accumulation over rows ----------
__global__ void __launch_bounds__(256)
bn_stats_kernel(const float* __restrict__ x, const float* __restrict__ bias,
                float* __restrict__ stats, int N, int C, int rowsPerBlk) {
  int c = threadIdx.x;
  if (c >= C) return;
  int r0 = blockIdx.x * rowsPerBlk;
  int r1 = r0 + rowsPerBlk; if (r1 > N) r1 = N;
  float b = bias[c];
  float s = 0.f, ss = 0.f;
  for (int r = r0; r < r1; ++r) {
    float v = x[(size_t)r * C + c] + b;
    s += v;
    ss = fmaf(v, v, ss);
  }
  atomicAdd(&stats[c], s);
  atomicAdd(&stats[256 + c], ss);
}

// ---------- fold bias/BN into per-channel scale+shift ----------
__global__ void bn_finalize_kernel(float* __restrict__ stats,
                                   const float* __restrict__ bias,
                                   const float* __restrict__ gamma,
                                   const float* __restrict__ beta,
                                   int N, int C) {
  int c = blockIdx.x * blockDim.x + threadIdx.x;
  if (c >= C) return;
  float invN = 1.f / (float)N;
  float mu = stats[c] * invN;
  float var = stats[256 + c] * invN - mu * mu;
  float scale = gamma[c] * rsqrtf(var + BN_EPS);
  float shift = (bias[c] - mu) * scale + beta[c];
  stats[c] = scale;
  stats[256 + c] = shift;
}

// ---------- apply BN + ELU in place ----------
__global__ void bn_apply_kernel(float* __restrict__ x, const float* __restrict__ stats,
                                int n, int C) {
  int t = blockIdx.x * blockDim.x + threadIdx.x;
  if (t >= n) return;
  int c = t % C;
  float y = fmaf(x[t], stats[c], stats[256 + c]);
  x[t] = (y > 0.f) ? y : expm1f(y);
}

// ---------- fused MLP head + log-softmax ----------
__global__ void mlp_head_kernel(const float* __restrict__ x,
                                const float* __restrict__ Wm, const float* __restrict__ bm,
                                const float* __restrict__ Wc, const float* __restrict__ bc,
                                float* __restrict__ logits, float* __restrict__ emb, int N) {
  int n = blockIdx.x * blockDim.x + threadIdx.x;
  if (n >= N) return;
  const float* xr = x + (size_t)n * 32;
  float h[16];
#pragma unroll
  for (int j = 0; j < 16; ++j) {
    float s = bm[j];
#pragma unroll
    for (int k = 0; k < 32; ++k) s = fmaf(xr[k], Wm[k * 16 + j], s);
    h[j] = fmaxf(s, 0.f);
    emb[(size_t)n * 16 + j] = h[j];
  }
  float l[8], mx = -3.4e38f;
#pragma unroll
  for (int o = 0; o < 8; ++o) {
    float s = bc[o];
#pragma unroll
    for (int j = 0; j < 16; ++j) s = fmaf(h[j], Wc[j * 8 + o], s);
    l[o] = s;
    mx = fmaxf(mx, s);
  }
  float se = 0.f;
#pragma unroll
  for (int o = 0; o < 8; ++o) se += __expf(l[o] - mx);
  float lse = __logf(se);
#pragma unroll
  for (int o = 0; o < 8; ++o) logits[(size_t)n * 8 + o] = l[o] - mx - lse;
}

// ---------- host launch ----------
static inline int gridFor(long long n, int bs) { return (int)((n + bs - 1) / bs); }

extern "C" void kernel_launch(void* const* d_in, const int* in_sizes, int n_in,
                              void* d_out, int out_size, void* d_ws, size_t ws_size,
                              hipStream_t stream) {
  const int N = in_sizes[0] / 128;      // 50000
  const int E = in_sizes[1] / 2;        // 800000

  const float* x0   = (const float*)d_in[0];
  const int*   eidx = (const int*)d_in[1];
  const int* esrc = eidx;
  const int* edst = eidx + E;

  const float *Ws[5], *asr[5], *adt[5], *bias[5], *gam[5], *bet[5];
  for (int i = 0; i < 5; ++i) {
    Ws[i]   = (const float*)d_in[2 + i];
    asr[i]  = (const float*)d_in[7 + i];
    adt[i]  = (const float*)d_in[12 + i];
    bias[i] = (const float*)d_in[17 + i];
    gam[i]  = (const float*)d_in[22 + i];
    bet[i]  = (const float*)d_in[27 + i];
  }
  const float* Wm = (const float*)d_in[32];
  const float* bm = (const float*)d_in[33];
  const float* Wc = (const float*)d_in[34];
  const float* bc = (const float*)d_in[35];

  // workspace carve (256B aligned)
  char* p = (char*)d_ws;
  auto carve = [&](size_t bytes) { char* r = p; p += (bytes + 255) & ~(size_t)255; return (void*)r; };
  float*        bufA  = (float*)carve((size_t)N * 256 * 4);          // GEMM out (xw)
  float*        bufB  = (float*)carve((size_t)N * 256 * 4);          // aggregated / next x
  unsigned int* Ap    = (unsigned int*)carve((size_t)N * 128 * 4);   // packed A frags (bf16 pairs)
  unsigned int* Bp    = (unsigned int*)carve((size_t)16 * 8 * 256 * 4); // packed B frags
  float*        es    = (float*)carve((size_t)N * 8 * 4);
  float*        ed    = (float*)carve((size_t)N * 8 * 4);
  unsigned int* mord  = (unsigned int*)carve((size_t)N * 8 * 4);
  float*        denom = (float*)carve((size_t)N * 8 * 4);
  float*        stats = (float*)carve(512 * 4);

  const int BS = 256;
  const int ROWS = 128;  // rows per bn_stats block
  const float* xin = x0;

  for (int i = 0; i < 5; ++i) {
    const int din = H_DIN[i], dout = H_DOUT[i], H = H_NHEADS[i];
    const int ksteps = din >> 5;
    const int nh  = N * H;
    const int eh  = (E + N) * H;
    const int nel = N * dout;

    int aTot = (N >> 4) * ksteps * 256;
    int bTot = (dout >> 4) * ksteps * 256;
    pack_b_kernel<<<gridFor(bTot, BS), BS, 0, stream>>>(Ws[i], Bp, din, dout);
    pack_a_kernel<<<gridFor(aTot, BS), BS, 0, stream>>>(xin, Ap, N, din);

    int waves = (((N >> 4) + 3) >> 2) * (dout >> 4);
    gemm_wmma4<<<gridFor((long long)waves * 32, BS), BS, 0, stream>>>(Ap, Bp, bufA, N, din, dout);

    attn_coef_kernel<<<gridFor(nh, BS), BS, 0, stream>>>(bufA, asr[i], adt[i], es, ed, N, H);

    zero_kernel<<<gridFor(nh, BS), BS, 0, stream>>>((float*)mord, nh);
    zero_kernel<<<gridFor(nh, BS), BS, 0, stream>>>(denom, nh);
    zero_kernel<<<gridFor(nel, BS), BS, 0, stream>>>(bufB, nel);
    zero_kernel<<<gridFor(512, BS), BS, 0, stream>>>(stats, 512);

    edge_max_kernel<<<gridFor(eh, BS), BS, 0, stream>>>(esrc, edst, es, ed, mord, E, N, H);
    edge_sum_kernel<<<gridFor(eh, BS), BS, 0, stream>>>(esrc, edst, es, ed, mord, denom, E, N, H);
    edge_scatter_kernel<<<gridFor((long long)eh * 8, BS), BS, 0, stream>>>(
        esrc, edst, es, ed, mord, denom, bufA, bufB, E, N, H);

    bn_stats_kernel<<<gridFor(N, ROWS), BS, 0, stream>>>(bufB, bias[i], stats, N, dout, ROWS);
    bn_finalize_kernel<<<1, 256, 0, stream>>>(stats, bias[i], gam[i], bet[i], N, dout);
    bn_apply_kernel<<<gridFor(nel, BS), BS, 0, stream>>>(bufB, stats, nel, dout);
    // residual never applies (out_dim != in_dim for all layers)
    xin = bufB;
  }

  // outputs: [N*8 log-softmax logits][N*16 emb]
  float* out_logits = (float*)d_out;
  float* out_emb    = (float*)d_out + (size_t)N * 8;
  mlp_head_kernel<<<gridFor(N, BS), BS, 0, stream>>>(bufB, Wm, bm, Wc, bc, out_logits, out_emb, N);
}
